// MHC_62603443306930
// MI455X (gfx1250) — matrix operand order repaired and checked by reading, not compiled, not run on hardware
//
#include <hip/hip_runtime.h>

#define D_MODEL   1024
#define NUM_HEADS 16
#define HEAD_DIM  64
#define BATCH     2
#define SEQ       2048
#define MROWS     (BATCH * SEQ)   // 4096

typedef __attribute__((ext_vector_type(16))) __bf16 bf16x16;
typedef __attribute__((ext_vector_type(8)))  __bf16 bf16x8;
typedef __attribute__((ext_vector_type(8)))  float  f32x8;

union AB16 { bf16x16 v; bf16x8 h[2]; };

__device__ __forceinline__ __bf16 f2bf(float f) {
    unsigned u = __builtin_bit_cast(unsigned, f);
    u += 0x7FFFu + ((u >> 16) & 1u);               // round-to-nearest-even
    unsigned short hs = (unsigned short)(u >> 16);
    return __builtin_bit_cast(__bf16, hs);
}

__device__ __forceinline__ f32x8 wmma_bf16(const AB16& a, const AB16& b, f32x8 c) {
    return __builtin_amdgcn_wmma_f32_16x16x32_bf16(
        /*neg_a*/false, a.v, /*neg_b*/false, b.v,
        /*c_mod*/(short)0, c, /*reuse_a*/false, /*reuse_b*/false);
}

__device__ __forceinline__ bf16x8 cvt8(float4 f0, float4 f1) {
    bf16x8 t;
    t[0]=f2bf(f0.x); t[1]=f2bf(f0.y); t[2]=f2bf(f0.z); t[3]=f2bf(f0.w);
    t[4]=f2bf(f1.x); t[5]=f2bf(f1.y); t[6]=f2bf(f1.z); t[7]=f2bf(f1.w);
    return t;
}

// ---------------------------------------------------------------------------
// Generic GEMM: C[M][N] = A[M][K] @ B[N][K]^T + bias[N]
// BM=256 (8 waves x 32 rows), BN=64, BK=32. Each wave holds TWO A fragments
// in registers and reuses each B fragment for both row-tiles:
// 4 A-loads + 8 B-loads -> 8 WMMAs per K-step (1.5 ds_load_b128 per WMMA).
// Register-prefetch software pipeline hides global latency behind WMMAs.
// ---------------------------------------------------------------------------
template<bool A_IS_BF16, bool STORE_BF16>
__global__ __launch_bounds__(256)
void wmma_gemm_bias(const void* __restrict__ Aany, const float* __restrict__ Bw,
                    const float* __restrict__ bias, void* __restrict__ Cany,
                    int M, int N, int K)
{
    constexpr int BM = 256, BN = 64, BK = 32, LDA = 40, LDB = 40;
    __shared__ __align__(16) __bf16 Alds[BM * LDA];
    __shared__ __align__(16) __bf16 Blds[BN * LDB];

    const int tid  = threadIdx.x;
    const int lane = tid & 31;
    const int wave = tid >> 5;
    const int ln16 = lane & 15;
    const int half = lane >> 4;
    const int rowBase = blockIdx.x * BM;
    const int colBase = blockIdx.y * BN;

    const float*  Af = (const float*)Aany;
    const __bf16* Ab = (const __bf16*)Aany;

    // staging: thread covers rows rA and rA+128, 16 elems each
    const int rA  = tid >> 1, kbA = (tid & 1) * 16;
    const int nB  = tid >> 2, kbB = (tid & 3) * 8;

    float4 fa[8], fb0, fb1;
    uint4  ua[4];

    auto loadTile = [&](int kc) {
        if (A_IS_BF16) {
            const uint4* s0 = (const uint4*)(Ab + (size_t)(rowBase + rA) * K + kc + kbA);
            const uint4* s1 = (const uint4*)(Ab + (size_t)(rowBase + rA + 128) * K + kc + kbA);
            ua[0] = s0[0]; ua[1] = s0[1]; ua[2] = s1[0]; ua[3] = s1[1];
        } else {
            const float4* s0 = (const float4*)(Af + (size_t)(rowBase + rA) * K + kc + kbA);
            const float4* s1 = (const float4*)(Af + (size_t)(rowBase + rA + 128) * K + kc + kbA);
            fa[0] = s0[0]; fa[1] = s0[1]; fa[2] = s0[2]; fa[3] = s0[3];
            fa[4] = s1[0]; fa[5] = s1[1]; fa[6] = s1[2]; fa[7] = s1[3];
        }
        const float4* sb = (const float4*)(Bw + (size_t)(colBase + nB) * K + kc + kbB);
        fb0 = sb[0]; fb1 = sb[1];
    };
    auto storeTile = [&]() {
        if (A_IS_BF16) {
            *(bf16x8*)&Alds[rA * LDA + kbA]             = __builtin_bit_cast(bf16x8, ua[0]);
            *(bf16x8*)&Alds[rA * LDA + kbA + 8]         = __builtin_bit_cast(bf16x8, ua[1]);
            *(bf16x8*)&Alds[(rA + 128) * LDA + kbA]     = __builtin_bit_cast(bf16x8, ua[2]);
            *(bf16x8*)&Alds[(rA + 128) * LDA + kbA + 8] = __builtin_bit_cast(bf16x8, ua[3]);
        } else {
            *(bf16x8*)&Alds[rA * LDA + kbA]             = cvt8(fa[0], fa[1]);
            *(bf16x8*)&Alds[rA * LDA + kbA + 8]         = cvt8(fa[2], fa[3]);
            *(bf16x8*)&Alds[(rA + 128) * LDA + kbA]     = cvt8(fa[4], fa[5]);
            *(bf16x8*)&Alds[(rA + 128) * LDA + kbA + 8] = cvt8(fa[6], fa[7]);
        }
        *(bf16x8*)&Blds[nB * LDB + kbB] = cvt8(fb0, fb1);
    };

    f32x8 acc[2][4] = {};
    loadTile(0);

    for (int kc = 0; kc < K; kc += BK) {
        storeTile();                          // consumes prefetch (waits loads)
        __syncthreads();
        if (kc + BK < K) loadTile(kc + BK);   // in flight during WMMAs below

        AB16 a0, a1;
        const int m0 = wave * 32 + ln16;
        const int m1 = m0 + 16;
        a0.h[0] = *(const bf16x8*)&Alds[m0 * LDA + half * 8];
        a0.h[1] = *(const bf16x8*)&Alds[m0 * LDA + 16 + half * 8];
        a1.h[0] = *(const bf16x8*)&Alds[m1 * LDA + half * 8];
        a1.h[1] = *(const bf16x8*)&Alds[m1 * LDA + 16 + half * 8];
        #pragma unroll
        for (int nt = 0; nt < 4; nt++) {
            AB16 b;
            const int n = nt * 16 + ln16;
            b.h[0] = *(const bf16x8*)&Blds[n * LDB + half * 8];
            b.h[1] = *(const bf16x8*)&Blds[n * LDB + 16 + half * 8];
            acc[0][nt] = wmma_bf16(a0, b, acc[0][nt]);
            acc[1][nt] = wmma_bf16(a1, b, acc[1][nt]);
        }
        __syncthreads();
    }

    #pragma unroll
    for (int mt = 0; mt < 2; mt++) {
        #pragma unroll
        for (int nt = 0; nt < 4; nt++) {
            const int n = colBase + nt * 16 + ln16;
            const float bv = bias ? bias[n] : 0.0f;
            #pragma unroll
            for (int r = 0; r < 8; r++) {
                const int mrow = rowBase + wave * 32 + mt * 16 + r + 8 * half;
                const float v = acc[mt][nt][r] + bv;
                if (STORE_BF16) ((__bf16*)Cany)[(size_t)mrow * N + n] = f2bf(v);
                else            ((float*) Cany)[(size_t)mrow * N + n] = v;
            }
        }
    }
}

// ---------------------------------------------------------------------------
// Conv1d (kernel=3, pad=1) over channels: 3 shifted GEMMs sharing one
// (256+2)-row LDS A tile; same 32-rows-per-wave + register-prefetch scheme.
// ---------------------------------------------------------------------------
__global__ __launch_bounds__(256)
void wmma_conv3_bias(const __bf16* __restrict__ A, const float* __restrict__ W,
                     const float* __restrict__ bias, __bf16* __restrict__ C,
                     int M, int N, int K, int Sseq)
{
    constexpr int BM = 256, BN = 64, BK = 32, LDA = 40, LDB = 40;
    __shared__ __align__(16) __bf16 Alds[(BM + 2) * LDA];
    __shared__ __align__(16) __bf16 Blds[3][BN * LDB];

    const int tid  = threadIdx.x;
    const int lane = tid & 31;
    const int wave = tid >> 5;
    const int ln16 = lane & 15;
    const int half = lane >> 4;
    const int rowBase = blockIdx.x * BM;
    const int colBase = blockIdx.y * BN;
    const int lo = (rowBase / Sseq) * Sseq;    // batch window (tile-aligned)
    const int hi = lo + Sseq;

    const __bf16 zb = f2bf(0.0f);

    // A: rows tid>>1, 128+(tid>>1), 256+(tid>>1) (last: tid<4 only)
    const int rA = tid >> 1, kbA = (tid & 1) * 16;
    const bool hasA2 = (tid < 4);
    const int nB = tid >> 2, kbB = (tid & 3) * 8;

    bf16x8 pa[3][2];
    float  pb[3][8];

    auto loadSeg = [&](int s, int row, int kc, bool en) {
        bf16x8 z = {zb,zb,zb,zb,zb,zb,zb,zb};
        pa[s][0] = z; pa[s][1] = z;
        const int gr = rowBase - 1 + row;
        if (en && gr >= lo && gr < hi) {
            const bf16x8* sp = (const bf16x8*)(A + (size_t)gr * K + kc + kbA);
            pa[s][0] = sp[0]; pa[s][1] = sp[1];
        }
    };
    auto loadTile = [&](int kc) {
        loadSeg(0, rA,       kc, true);
        loadSeg(1, rA + 128, kc, true);
        loadSeg(2, rA + 256, kc, hasA2);
        #pragma unroll
        for (int kk = 0; kk < 3; kk++) {
            const float* src = W + (size_t)(colBase + nB) * (3 * K) + (size_t)(kc + kbB) * 3 + kk;
            #pragma unroll
            for (int e = 0; e < 8; e++) pb[kk][e] = src[e * 3];
        }
    };
    auto storeTile = [&]() {
        *(bf16x8*)&Alds[rA * LDA + kbA]             = pa[0][0];
        *(bf16x8*)&Alds[rA * LDA + kbA + 8]         = pa[0][1];
        *(bf16x8*)&Alds[(rA + 128) * LDA + kbA]     = pa[1][0];
        *(bf16x8*)&Alds[(rA + 128) * LDA + kbA + 8] = pa[1][1];
        if (hasA2) {
            *(bf16x8*)&Alds[(rA + 256) * LDA + kbA]     = pa[2][0];
            *(bf16x8*)&Alds[(rA + 256) * LDA + kbA + 8] = pa[2][1];
        }
        #pragma unroll
        for (int kk = 0; kk < 3; kk++) {
            bf16x8 t;
            #pragma unroll
            for (int e = 0; e < 8; e++) t[e] = f2bf(pb[kk][e]);
            *(bf16x8*)&Blds[kk][nB * LDB + kbB] = t;
        }
    };

    f32x8 acc[2][4] = {};
    loadTile(0);

    for (int kc = 0; kc < K; kc += BK) {
        storeTile();
        __syncthreads();
        if (kc + BK < K) loadTile(kc + BK);

        #pragma unroll
        for (int kk = 0; kk < 3; kk++) {
            AB16 a0, a1;
            const int m0 = wave * 32 + ln16 + kk;       // shifted rows
            const int m1 = m0 + 16;
            a0.h[0] = *(const bf16x8*)&Alds[m0 * LDA + half * 8];
            a0.h[1] = *(const bf16x8*)&Alds[m0 * LDA + 16 + half * 8];
            a1.h[0] = *(const bf16x8*)&Alds[m1 * LDA + half * 8];
            a1.h[1] = *(const bf16x8*)&Alds[m1 * LDA + 16 + half * 8];
            #pragma unroll
            for (int nt = 0; nt < 4; nt++) {
                AB16 b;
                const int n = nt * 16 + ln16;
                b.h[0] = *(const bf16x8*)&Blds[kk][n * LDB + half * 8];
                b.h[1] = *(const bf16x8*)&Blds[kk][n * LDB + 16 + half * 8];
                acc[0][nt] = wmma_bf16(a0, b, acc[0][nt]);
                acc[1][nt] = wmma_bf16(a1, b, acc[1][nt]);
            }
        }
        __syncthreads();
    }

    #pragma unroll
    for (int mt = 0; mt < 2; mt++) {
        #pragma unroll
        for (int nt = 0; nt < 4; nt++) {
            const int n = colBase + nt * 16 + ln16;
            const float bv = bias[n];
            #pragma unroll
            for (int r = 0; r < 8; r++) {
                const int mrow = rowBase + wave * 32 + mt * 16 + r + 8 * half;
                C[(size_t)mrow * N + n] = f2bf(acc[mt][nt][r] + bv);
            }
        }
    }
}

// ---------------------------------------------------------------------------
// Causal flash attention, bf16 in / bf16 out, f32 accumulate.
// Each wave owns 32 query rows (2 row-tiles): K and V fragments are loaded
// once per chunk and reused by both row-tiles (1 ds_load_b128 per WMMA).
// K chunk staging via GLOBAL_LOAD_ASYNC_TO_LDS_B128 (ASYNCcnt path).
// ---------------------------------------------------------------------------
#define KLDS_BYTES (32 * 72 * 2)
#define VT_BYTES   (64 * 40 * 2)
#define PLDS_BYTES (8 * 16 * 40 * 2)

__global__ __launch_bounds__(256)
void wmma_flash_attn(const __bf16* __restrict__ Q, int qStride,
                     const __bf16* __restrict__ Kp, const __bf16* __restrict__ Vp,
                     int kvStride,
                     __bf16* __restrict__ O, int oStride,
                     int S, float scale)
{
    __shared__ __align__(16) char smem[KLDS_BYTES + VT_BYTES + PLDS_BYTES];
    __bf16* Klds = (__bf16*)smem;                              // [32][72]
    __bf16* Vt   = (__bf16*)(smem + KLDS_BYTES);               // [64][40]
    __bf16* Plds = (__bf16*)(smem + KLDS_BYTES + VT_BYTES);    // [8][16*40]

    const int tid  = threadIdx.x;
    const int lane = tid & 31;
    const int wave = tid >> 5;
    const int ln16 = lane & 15;
    const int half = lane >> 4;
    const int bh = blockIdx.y;
    const int b  = bh / NUM_HEADS;
    const int h  = bh % NUM_HEADS;
    const int qBase = blockIdx.x * 256;
    const int q0 = qBase + wave * 32;                 // 32 rows per wave
    const float L2E = 1.44269504f;

    // Q fragments: 2 row-tiles x 2 K-chunks, register-resident
    AB16 qf[2][2];
    #pragma unroll
    for (int rt = 0; rt < 2; rt++) {
        const __bf16* qrow = Q + ((size_t)b * S + (q0 + rt * 16 + ln16)) * qStride + h * HEAD_DIM;
        #pragma unroll
        for (int c = 0; c < 2; c++) {
            qf[rt][c].h[0] = *(const bf16x8*)(qrow + c * 32 + half * 8);
            qf[rt][c].h[1] = *(const bf16x8*)(qrow + c * 32 + 16 + half * 8);
        }
    }

    const int kKey = tid >> 3, kDseg = (tid & 7) * 8;          // K: 16B async
    const unsigned kLdsOff = (unsigned)((kKey * 72 + kDseg) * 2);

    float mrow[2][8], lrow[2][8];
    #pragma unroll
    for (int rt = 0; rt < 2; rt++)
        #pragma unroll
        for (int r = 0; r < 8; r++) { mrow[rt][r] = -3.0e38f; lrow[rt][r] = 0.0f; }
    f32x8 Ov[2][4] = {};

    const int jEnd = qBase + 256;
    for (int j0 = 0; j0 < jEnd; j0 += 32) {
        // ---- K chunk (32 keys x 64 d): async global -> LDS, bypass VGPRs ----
        {
            const __bf16* gp = Kp + ((size_t)b * S + j0 + kKey) * kvStride + h * HEAD_DIM + kDseg;
            unsigned long long ga = (unsigned long long)(uintptr_t)gp;
            asm volatile("global_load_async_to_lds_b128 %0, %1, off"
                         :: "v"(kLdsOff), "v"(ga) : "memory");
        }
        // ---- V chunk transposed into Vt[d][key] (element-wise) ----
        #pragma unroll
        for (int e = 0; e < 8; e++) {
            const int idx = tid + e * 256;
            const int key = idx >> 6, d = idx & 63;
            Vt[d * 40 + key] = Vp[((size_t)b * S + j0 + key) * kvStride + h * HEAD_DIM + d];
        }
        asm volatile("s_wait_asynccnt 0x0" ::: "memory");
        __syncthreads();

        if (j0 <= q0 + 31) {               // uniform per wave; causal skip
            // ---- scores: 2 row-tiles x 2 key-subtiles, kf reused per rt ----
            f32x8 sc[2][2] = {};
            #pragma unroll
            for (int t = 0; t < 2; t++) {
                #pragma unroll
                for (int c = 0; c < 2; c++) {
                    AB16 kf;
                    const int key = t * 16 + ln16;
                    kf.h[0] = *(const bf16x8*)&Klds[key * 72 + c * 32 + half * 8];
                    kf.h[1] = *(const bf16x8*)&Klds[key * 72 + c * 32 + 16 + half * 8];
                    #pragma unroll
                    for (int rt = 0; rt < 2; rt++)
                        sc[rt][t] = wmma_bf16(qf[rt][c], kf, sc[rt][t]);
                }
            }
            // ---- per row-tile: mask + online softmax + P fragment ----
            float alpha[2][8];
            AB16 pf[2];
            __bf16* pw = Plds + wave * (16 * 40);
            const int k0 = j0 + ln16, k1 = j0 + 16 + ln16;
            #pragma unroll
            for (int rt = 0; rt < 2; rt++) {
                float p0v[8], p1v[8];
                #pragma unroll
                for (int r = 0; r < 8; r++) {
                    const int qrow = q0 + rt * 16 + r + 8 * half;
                    float s0 = sc[rt][0][r] * scale;
                    float s1 = sc[rt][1][r] * scale;
                    if (k0 > qrow) s0 = -3.0e38f;
                    if (k1 > qrow) s1 = -3.0e38f;
                    float mx = fmaxf(s0, s1);
                    #pragma unroll
                    for (int off = 1; off < 16; off <<= 1)
                        mx = fmaxf(mx, __shfl_xor(mx, off, 32));
                    const float mnew = fmaxf(mrow[rt][r], mx);
                    const float a  = exp2f((mrow[rt][r] - mnew) * L2E);
                    const float p0 = exp2f((s0 - mnew) * L2E);
                    const float p1 = exp2f((s1 - mnew) * L2E);
                    float ps = p0 + p1;
                    #pragma unroll
                    for (int off = 1; off < 16; off <<= 1)
                        ps += __shfl_xor(ps, off, 32);
                    lrow[rt][r] = lrow[rt][r] * a + ps;
                    mrow[rt][r] = mnew;
                    alpha[rt][r] = a; p0v[r] = p0; p1v[r] = p1;
                }
                // C-layout -> per-wave LDS scratch -> A-layout fragment
                #pragma unroll
                for (int r = 0; r < 8; r++) {
                    pw[(r + 8 * half) * 40 + ln16]      = f2bf(p0v[r]);
                    pw[(r + 8 * half) * 40 + 16 + ln16] = f2bf(p1v[r]);
                }
                pf[rt].h[0] = *(const bf16x8*)&pw[ln16 * 40 + half * 8];
                pf[rt].h[1] = *(const bf16x8*)&pw[ln16 * 40 + 16 + half * 8];
            }
            // ---- P @ V: vf loaded once per d-block, reused by both rt ----
            #pragma unroll
            for (int dblk = 0; dblk < 4; dblk++) {
                AB16 vf;
                const int d = dblk * 16 + ln16;
                vf.h[0] = *(const bf16x8*)&Vt[d * 40 + half * 8];
                vf.h[1] = *(const bf16x8*)&Vt[d * 40 + 16 + half * 8];
                #pragma unroll
                for (int rt = 0; rt < 2; rt++) {
                    #pragma unroll
                    for (int r = 0; r < 8; r++) Ov[rt][dblk][r] *= alpha[rt][r];
                    Ov[rt][dblk] = wmma_bf16(pf[rt], vf, Ov[rt][dblk]);
                }
            }
        }
        __syncthreads();
    }

    // ---- normalize and store bf16 ----
    #pragma unroll
    for (int rt = 0; rt < 2; rt++) {
        #pragma unroll
        for (int dblk = 0; dblk < 4; dblk++) {
            #pragma unroll
            for (int r = 0; r < 8; r++) {
                const int row = q0 + rt * 16 + r + 8 * half;
                const float v = Ov[rt][dblk][r] / lrow[rt][r];
                O[((size_t)b * S + row) * oStride + h * HEAD_DIM + dblk * 16 + ln16] = f2bf(v);
            }
        }
    }
}

// ---------------------------------------------------------------------------
extern "C" void kernel_launch(void* const* d_in, const int* in_sizes, int n_in,
                              void* d_out, int out_size, void* d_ws, size_t ws_size,
                              hipStream_t stream)
{
    const float* x      = (const float*)d_in[0];
    const float* Wqkv_w = (const float*)d_in[1];
    const float* Wqkv_b = (const float*)d_in[2];
    const float* conv_w = (const float*)d_in[3];
    const float* conv_b = (const float*)d_in[4];
    const float* out_w  = (const float*)d_in[5];
    const float* out_b  = (const float*)d_in[6];

    // workspace: all intermediates in bf16 (keeps everything inside L2)
    __bf16* qkv  = (__bf16*)d_ws;                         // 4096 x 3072
    __bf16* ctx  = qkv  + (size_t)MROWS * 3 * D_MODEL;    // 4096 x 1024
    __bf16* q2   = ctx  + (size_t)MROWS * D_MODEL;        // 4096 x 1024
    __bf16* ctx2 = q2   + (size_t)MROWS * D_MODEL;        // 4096 x 1024

    const dim3 blk(256);
    const float scale = 0.125f;   // 1/sqrt(64)

    // 1) QKV projection -> bf16
    wmma_gemm_bias<false, true><<<dim3(MROWS / 256, (3 * D_MODEL) / 64), blk, 0, stream>>>(
        x, Wqkv_w, Wqkv_b, qkv, MROWS, 3 * D_MODEL, D_MODEL);

    // 2) causal attention #1: q,k,v strided inside qkv rows (stride 3*D)
    wmma_flash_attn<<<dim3(SEQ / 256, BATCH * NUM_HEADS), blk, 0, stream>>>(
        qkv, 3 * D_MODEL,
        qkv + D_MODEL, qkv + 2 * D_MODEL, 3 * D_MODEL,
        ctx, D_MODEL, SEQ, scale);

    // 3) conv1d (k=3, pad=1) over channels -> q2 (bf16)
    wmma_conv3_bias<<<dim3(MROWS / 256, D_MODEL / 64), blk, 0, stream>>>(
        ctx, conv_w, conv_b, q2, MROWS, D_MODEL, D_MODEL, SEQ);

    // 4) causal attention #2 with q2, same k/v
    wmma_flash_attn<<<dim3(SEQ / 256, BATCH * NUM_HEADS), blk, 0, stream>>>(
        q2, D_MODEL,
        qkv + D_MODEL, qkv + 2 * D_MODEL, 3 * D_MODEL,
        ctx2, D_MODEL, SEQ, scale);

    // 5) output projection -> f32 d_out
    wmma_gemm_bias<true, false><<<dim3(MROWS / 256, D_MODEL / 64), blk, 0, stream>>>(
        ctx2, out_w, out_b, (float*)d_out, MROWS, D_MODEL, D_MODEL);
}
